// DimeNet_48962627175131
// MI455X (gfx1250) — compile-verified, matrix operand order checked by reading, not run on hardware
//
#include <hip/hip_runtime.h>
#include <math.h>
#include <stdint.h>

// ---------------- constants matching the reference ----------------
#define EMB   128
#define NRAD  6
#define NSPH  7
#define NSBF  42      // NSPH * NRAD
#define NBIL  8
#define NBLK  4
#define NTGT  12
#define CUTOFF_R 5.0f

typedef unsigned short u16;
typedef unsigned int   u32;
typedef __bf16 bf16_t;
typedef bf16_t v16bf __attribute__((ext_vector_type(16)));
typedef float  v8f   __attribute__((ext_vector_type(8)));

// spherical Bessel zeros j_l, l=0..6, first 6 each
__device__ __constant__ float c_sbf_zeros[NSPH][NRAD] = {
  {3.141592653589793f, 6.283185307179586f, 9.424777960769379f, 12.566370614359172f, 15.707963267948966f, 18.849555921538759f},
  {4.493409457909064f, 7.725251836937707f, 10.904121659428899f, 14.066193912831473f, 17.220755271930768f, 20.371302959287563f},
  {5.763459196894550f, 9.095011330476355f, 12.322940970566582f, 15.514603010886749f, 18.689036355362822f, 21.853874222709714f},
  {6.987932000500519f, 10.417118547379365f, 13.698023153249246f, 16.923621285214318f, 20.121806174454748f, 23.304246988939617f},
  {8.182561452571243f, 11.704907154571369f, 15.039664707616520f, 18.301255959541885f, 21.525417733399979f, 24.727565547835540f},
  {9.355812111042747f, 12.966530172775054f, 16.354709639371686f, 19.653152101821306f, 22.904550647902580f, 26.127750137074675f},
  {10.512835408093997f, 14.207392458843459f, 17.647974941206350f, 20.983463068910367f, 24.262768364876424f, 27.507868401352864f}
};

// ---------------- small device helpers ----------------
__device__ __forceinline__ float bf2f(u16 h) {
  u32 u = ((u32)h) << 16;
  return __builtin_bit_cast(float, u);
}
__device__ __forceinline__ u16 f2bf(float f) {
  u32 u = __builtin_bit_cast(u32, f);
  u32 r = u + 0x7FFFu + ((u >> 16) & 1u);   // round-to-nearest-even
  return (u16)(r >> 16);
}
__device__ __forceinline__ bf16_t bfbits(u16 h) {
  return __builtin_bit_cast(bf16_t, h);
}
__device__ __forceinline__ float mishf(float x) {
  float sp = (x > 20.f) ? x : log1pf(expf(x));
  return x * tanhf(sp);
}
// WMMA 16x32 bf16 fragment K index for (vgpr v, lane-half hi, half p)
__device__ __forceinline__ int kmap(int v, int hi, int p) {
  return ((v < 4) ? 0 : 16) + (hi << 3) + ((v & 3) << 1) + p;
}

// --- CDNA5 async global->LDS copy (ASYNCcnt-tracked, §15.18.3 op 98) ---
__device__ __forceinline__ void async_copy16(u32 lds_off, const u16* g) {
  unsigned long long ga = (unsigned long long)(uintptr_t)g;
  asm volatile("global_load_async_to_lds_b128 %0, %1, off"
               :: "v"(lds_off), "v"(ga) : "memory");
}
__device__ __forceinline__ void async_wait0() {
  asm volatile("s_wait_asynccnt 0x0" ::: "memory");
}

// ---------------- generic bf16 WMMA GEMM (N fixed = 128) ----------------
// C[M,128] = [acc +] act(A[M,K] @ W[K,128] + bias)
// Each wave owns one 16-row strip and all 8 column tiles; the A fragment is
// reused across 8 WMMAs per K-step. WT is bf16 [128,K] (pre-transposed).
// For K==128 the whole 32KB weight is staged once per workgroup into LDS via
// the async-to-LDS path, so hot-loop B reads are ds_load instead of L2.
__global__ void __launch_bounds__(256)
k_wmma_gemm(const u16* __restrict__ A, const u16* __restrict__ WT,
            const float* __restrict__ bias, float* __restrict__ C,
            u16* __restrict__ Cbf, int M, int K, int act, int accumulate) {
  __shared__ u16 sWT[EMB * EMB];             // 32 KB
  const int tid  = threadIdx.x;
  const int lane = tid & 31;
  const int wave = tid >> 5;
  const bool useLds = (K == EMB);
  if (useLds) {
    for (int ch = tid; ch < EMB * EMB / 8; ch += 256) {
      int i = ch * 8;                         // 8 u16 = 16 bytes per issue
      async_copy16((u32)(uintptr_t)&sWT[i], WT + i);
    }
    async_wait0();
    __syncthreads();
  }
  const int m0 = (blockIdx.x * 8 + wave) << 4;
  if (m0 >= M) return;                        // after the only barrier
  const int mr = lane & 15, hi = lane >> 4;
  int arowi = m0 + mr; if (arowi > M - 1) arowi = M - 1;
  const u16* __restrict__ arow = A + (size_t)arowi * K;
  v8f c[8];
#pragma unroll
  for (int nt = 0; nt < 8; ++nt) c[nt] = v8f{};
  for (int k0 = 0; k0 < K; k0 += 32) {
    __builtin_prefetch(arow + k0 + 64, 0, 0);
    v16bf a;
#pragma unroll
    for (int v = 0; v < 8; ++v)
#pragma unroll
      for (int p = 0; p < 2; ++p)
        a[v * 2 + p] = bfbits(arow[k0 + kmap(v, hi, p)]);
    if (useLds) {
#pragma unroll
      for (int nt = 0; nt < 8; ++nt) {
        const int bb = (nt * 16 + mr) * EMB + k0;
        v16bf b;
#pragma unroll
        for (int v = 0; v < 8; ++v)
#pragma unroll
          for (int p = 0; p < 2; ++p)
            b[v * 2 + p] = bfbits(sWT[bb + kmap(v, hi, p)]);
        c[nt] = __builtin_amdgcn_wmma_f32_16x16x32_bf16(false, a, false, b,
                                                        (short)0, c[nt],
                                                        false, false);
      }
    } else {
#pragma unroll
      for (int nt = 0; nt < 8; ++nt) {
        const u16* __restrict__ brow = WT + (size_t)(nt * 16 + mr) * K + k0;
        v16bf b;
#pragma unroll
        for (int v = 0; v < 8; ++v)
#pragma unroll
          for (int p = 0; p < 2; ++p)
            b[v * 2 + p] = bfbits(brow[kmap(v, hi, p)]);
        c[nt] = __builtin_amdgcn_wmma_f32_16x16x32_bf16(false, a, false, b,
                                                        (short)0, c[nt],
                                                        false, false);
      }
    }
  }
#pragma unroll
  for (int nt = 0; nt < 8; ++nt) {
    const int n = nt * 16 + mr;
    const float bv = bias ? bias[n] : 0.f;
#pragma unroll
    for (int r = 0; r < 8; ++r) {
      int m = m0 + r + (hi << 3);
      if (m >= M) continue;
      size_t idx = (size_t)m * EMB + n;
      float v = c[nt][r] + bv;
      if (act) v = mishf(v);
      if (accumulate) v += C[idx];
      C[idx] = v;
      if (Cbf) Cbf[idx] = f2bf(v);
    }
  }
}

// ---------------- fused bilinear ----------------
// out[id_reduce[w],i] += sum_j sum_l sbfp[w,j] * xkj[id_expand[w],l] * bilin[i,j,l]
// One wave per 16-triplet strip, all 8 column tiles. Each B_j (32KB) is staged
// into LDS per uniform j-iteration (async copy + barriers); all waves stay
// alive through the barriers, with stores guarded by validity.
__global__ void __launch_bounds__(256)
k_wmma_bilinear(const u16* __restrict__ xkj_bf, const float* __restrict__ sbf_p,
                const u16* __restrict__ bilinT, const int* __restrict__ id_expand,
                const int* __restrict__ id_reduce, float* __restrict__ out, int T) {
  __shared__ u16 sB[EMB * EMB];              // 32 KB
  const int tid  = threadIdx.x;
  const int lane = tid & 31;
  const int wave = tid >> 5;
  const int m0 = (blockIdx.x * 8 + wave) << 4;
  const int mr = lane & 15, hi = lane >> 4;
  int w = m0 + mr; if (w > T - 1) w = T - 1;
  const int e = id_expand[w];
  const u16* __restrict__ xrow = xkj_bf + (size_t)e * EMB;
  float s[NBIL];
#pragma unroll
  for (int j = 0; j < NBIL; ++j) s[j] = sbf_p[(size_t)w * NBIL + j];
  v8f c[8];
#pragma unroll
  for (int nt = 0; nt < 8; ++nt) c[nt] = v8f{};
  for (int j = 0; j < NBIL; ++j) {
    __syncthreads();                          // previous stage fully consumed
    {
      const u16* src = bilinT + (size_t)j * EMB * EMB;
      for (int ch = tid; ch < EMB * EMB / 8; ch += 256) {
        int i = ch * 8;
        async_copy16((u32)(uintptr_t)&sB[i], src + i);
      }
      async_wait0();
    }
    __syncthreads();
    const float sj = s[j];
#pragma unroll
    for (int k0 = 0; k0 < EMB; k0 += 32) {
      v16bf a;
#pragma unroll
      for (int v = 0; v < 8; ++v)
#pragma unroll
        for (int p = 0; p < 2; ++p)
          a[v * 2 + p] = bfbits(f2bf(bf2f(xrow[k0 + kmap(v, hi, p)]) * sj));
#pragma unroll
      for (int nt = 0; nt < 8; ++nt) {
        const int bb = (nt * 16 + mr) * EMB + k0;
        v16bf b;
#pragma unroll
        for (int v = 0; v < 8; ++v)
#pragma unroll
          for (int p = 0; p < 2; ++p)
            b[v * 2 + p] = bfbits(sB[bb + kmap(v, hi, p)]);
        c[nt] = __builtin_amdgcn_wmma_f32_16x16x32_bf16(false, a, false, b,
                                                        (short)0, c[nt],
                                                        false, false);
      }
    }
  }
#pragma unroll
  for (int nt = 0; nt < 8; ++nt) {
    const int n = nt * 16 + mr;
#pragma unroll
    for (int r = 0; r < 8; ++r) {
      int m = m0 + r + (hi << 3);
      if (m >= T) continue;
      int tgt = id_reduce[m];
      atomicAdd(&out[(size_t)tgt * EMB + n], c[nt][r]);
    }
  }
}

// ---------------- packing / elementwise helpers ----------------
__global__ void k_pack_wt(const float* __restrict__ W, u16* __restrict__ WT,
                          int K, int N) {
  int i = blockIdx.x * blockDim.x + threadIdx.x;
  if (i >= K * N) return;
  int n = i % N, k = i / N;
  WT[(size_t)n * K + k] = f2bf(W[i]);
}
// bilin one block: src [EMB][NBIL][EMB] -> dst [NBIL][EMB(i)][EMB(l)]
__global__ void k_pack_bilin(const float* __restrict__ src, u16* __restrict__ dst) {
  int idx = blockIdx.x * blockDim.x + threadIdx.x;
  if (idx >= EMB * NBIL * EMB) return;
  int l = idx % EMB;
  int j = (idx / EMB) % NBIL;
  int i = idx / (EMB * NBIL);
  dst[((size_t)j * EMB + i) * EMB + l] = f2bf(src[idx]);
}
__global__ void k_zero_f32(float* __restrict__ p, size_t n) {
  size_t i = (size_t)blockIdx.x * blockDim.x + threadIdx.x;
  if (i < n) p[i] = 0.f;
}
__global__ void k_f32_to_bf16(const float* __restrict__ a, u16* __restrict__ o, size_t n) {
  size_t i = (size_t)blockIdx.x * blockDim.x + threadIdx.x;
  if (i < n) o[i] = f2bf(a[i]);
}
__global__ void k_add_bf(const float* __restrict__ a, const float* __restrict__ b,
                         float* __restrict__ o, u16* __restrict__ obf, size_t n) {
  size_t i = (size_t)blockIdx.x * blockDim.x + threadIdx.x;
  if (i >= n) return;
  float v = a[i] + b[i];
  o[i] = v;
  obf[i] = f2bf(v);
}
__global__ void k_accum_bf(float* __restrict__ x, const float* __restrict__ d,
                           u16* __restrict__ xbf, size_t n) {
  size_t i = (size_t)blockIdx.x * blockDim.x + threadIdx.x;
  if (i >= n) return;
  float v = x[i] + d[i];
  x[i] = v;
  xbf[i] = f2bf(v);
}

// ---------------- norm table: (0.5 * j_{l+1}(z)^2)^{-1/2} ----------------
__global__ void k_init_norm(float* __restrict__ normtab) {
  int idx = threadIdx.x;
  if (idx >= NSPH * NRAD) return;
  int l = idx / NRAD;
  float z = c_sbf_zeros[l][idx % NRAD];
  float j0 = sinf(z) / z;
  float j1 = sinf(z) / (z * z) - cosf(z) / z;
  float jm = j0, jc = j1;
  for (int o = 2; o <= l + 1; ++o) {
    float t = (2.f * o - 1.f) / z * jc - jm;
    jm = jc; jc = t;
  }
  normtab[idx] = rsqrtf(0.5f * jc * jc);
}

// ---------------- edge basis: D, envelope, rbf [E,6], rbf_env [E,42] ----------------
__global__ void k_edge_basis(const float* __restrict__ R, const int* __restrict__ ei,
                             const int* __restrict__ ej, const float* __restrict__ freq,
                             const float* __restrict__ normtab, float* __restrict__ rbf,
                             float* __restrict__ rbf_env, int E) {
  int e = blockIdx.x * blockDim.x + threadIdx.x;
  if (e >= E) return;
  int i = ei[e], j = ej[e];
  float dx = R[i * 3 + 0] - R[j * 3 + 0];
  float dy = R[i * 3 + 1] - R[j * 3 + 1];
  float dz = R[i * 3 + 2] - R[j * 3 + 2];
  float D = sqrtf(fmaxf(dx * dx + dy * dy + dz * dz, 0.f));
  float x = D / CUTOFF_R;
  float env = 0.f;
  if (x < 1.f) {
    float x2 = x * x, x4 = x2 * x2, x5 = x4 * x, x6 = x5 * x, x7 = x6 * x;
    env = 1.f / x - 28.f * x5 + 48.f * x6 - 21.f * x7;  // p=6 envelope
  }
  for (int r = 0; r < NRAD; ++r)
    rbf[(size_t)e * NRAD + r] = env * sinf(freq[r] * x);
  for (int l = 0; l < NSPH; ++l) {
    for (int n = 0; n < NRAD; ++n) {
      float z = c_sbf_zeros[l][n] * x;
      float j0 = sinf(z) / z;
      float jl = j0;
      if (l >= 1) {
        float j1 = sinf(z) / (z * z) - cosf(z) / z;
        float jm = j0, jc = j1;
        for (int o = 2; o <= l; ++o) {
          float t = (2.f * o - 1.f) / z * jc - jm;
          jm = jc; jc = t;
        }
        jl = jc;
      }
      rbf_env[(size_t)e * NSBF + l * NRAD + n] = env * normtab[l * NRAD + n] * jl;
    }
  }
}

// ---------------- triplet: angle, Legendre, sbf [T,42] ----------------
__global__ void k_triplet_sbf(const float* __restrict__ R, const int* __restrict__ ti,
                              const int* __restrict__ tj, const int* __restrict__ tk,
                              const int* __restrict__ id_expand,
                              const float* __restrict__ rbf_env,
                              float* __restrict__ sbf, int T) {
  int w = blockIdx.x * blockDim.x + threadIdx.x;
  if (w >= T) return;
  int i = ti[w], j = tj[w], k = tk[w];
  float ax = R[j * 3 + 0] - R[i * 3 + 0];
  float ay = R[j * 3 + 1] - R[i * 3 + 1];
  float az = R[j * 3 + 2] - R[i * 3 + 2];
  float bx = R[k * 3 + 0] - R[i * 3 + 0];
  float by = R[k * 3 + 1] - R[i * 3 + 1];
  float bz = R[k * 3 + 2] - R[i * 3 + 2];
  float dot = ax * bx + ay * by + az * bz;
  float cx = ay * bz - az * by, cy = az * bx - ax * bz, cz = ax * by - ay * bx;
  float cr = sqrtf(cx * cx + cy * cy + cz * cz);
  float cth = cosf(atan2f(cr, dot));
  float P[NSPH];
  P[0] = 1.f; P[1] = cth;
  for (int l = 2; l < NSPH; ++l)
    P[l] = ((2.f * l - 1.f) * cth * P[l - 1] - (l - 1.f) * P[l - 2]) / (float)l;
  const float* __restrict__ re = rbf_env + (size_t)id_expand[w] * NSBF;
  for (int l = 0; l < NSPH; ++l) {
    float cb = sqrtf((2.f * l + 1.f) / (4.f * 3.14159265358979f)) * P[l];
    for (int n = 0; n < NRAD; ++n)
      sbf[(size_t)w * NSBF + l * NRAD + n] = re[l * NRAD + n] * cb;
  }
}

// ---------------- embedding feature build: bf16 [E, 3*EMB] ----------------
__global__ void k_embed_feat(const float* __restrict__ emb, const int* __restrict__ Z,
                             const int* __restrict__ ei, const int* __restrict__ ej,
                             const float* __restrict__ rbf, const float* __restrict__ Wrbf,
                             const float* __restrict__ brbf, u16* __restrict__ Aemb, int E) {
  int t = blockIdx.x * blockDim.x + threadIdx.x;
  if (t >= E * EMB) return;
  int c = t % EMB, e = t / EMB;
  float acc = brbf[c];
  for (int r = 0; r < NRAD; ++r) acc += rbf[(size_t)e * NRAD + r] * Wrbf[r * EMB + c];
  float rbfe = mishf(acc);
  int zi = Z[ei[e]], zj = Z[ej[e]];
  u16* row = Aemb + (size_t)e * (3 * EMB);
  row[c]           = f2bf(emb[(size_t)zi * EMB + c]);
  row[EMB + c]     = f2bf(emb[(size_t)zj * EMB + c]);
  row[2 * EMB + c] = f2bf(rbfe);
}

// ---------------- sbf_p = sbf @ Wsbf  [T,8] ----------------
__global__ void k_sbf_project(const float* __restrict__ sbf, const float* __restrict__ W,
                              float* __restrict__ sbf_p, int T) {
  int w = blockIdx.x * blockDim.x + threadIdx.x;
  if (w >= T) return;
  const float* srow = sbf + (size_t)w * NSBF;
  float acc[NBIL];
#pragma unroll
  for (int j = 0; j < NBIL; ++j) acc[j] = 0.f;
  for (int q = 0; q < NSBF; ++q) {
    float sv = srow[q];
#pragma unroll
    for (int j = 0; j < NBIL; ++j) acc[j] += sv * W[q * NBIL + j];
  }
#pragma unroll
  for (int j = 0; j < NBIL; ++j) sbf_p[(size_t)w * NBIL + j] = acc[j];
}

// ---------------- x_kj scale by (rbf @ Wrbf), write bf16 ----------------
__global__ void k_kj_scale(const float* __restrict__ rbf, const float* __restrict__ Wrbf,
                           const float* __restrict__ xkj, u16* __restrict__ xkj_bf, int E) {
  int t = blockIdx.x * blockDim.x + threadIdx.x;
  if (t >= E * EMB) return;
  int c = t % EMB, e = t / EMB;
  float s = 0.f;
  for (int r = 0; r < NRAD; ++r) s += rbf[(size_t)e * NRAD + r] * Wrbf[r * EMB + c];
  xkj_bf[t] = f2bf(xkj[t] * s);
}

// ---------------- output block gather: t_atom += (rbf@W) * x ----------------
__global__ void k_out_gather(const float* __restrict__ rbf, const float* __restrict__ Wrbf,
                             const float* __restrict__ x, const int* __restrict__ ei,
                             float* __restrict__ t_atom, int E) {
  int t = blockIdx.x * blockDim.x + threadIdx.x;
  if (t >= E * EMB) return;
  int c = t % EMB, e = t / EMB;
  float g = 0.f;
  for (int r = 0; r < NRAD; ++r) g += rbf[(size_t)e * NRAD + r] * Wrbf[r * EMB + c];
  atomicAdd(&t_atom[(size_t)ei[e] * EMB + c], g * x[t]);
}

// ---------------- final head: P[a] = t[a] @ Wf, molecule segment-sum ----------------
__global__ void k_final_out(const float* __restrict__ t, const float* __restrict__ Wf,
                            const int* __restrict__ batch_seg, float* __restrict__ out,
                            int natoms) {
  int idx = blockIdx.x * blockDim.x + threadIdx.x;
  if (idx >= natoms * NTGT) return;
  int tg = idx % NTGT, a = idx / NTGT;
  const float* row = t + (size_t)a * EMB;
  float acc = 0.f;
  for (int c = 0; c < EMB; ++c) acc += row[c] * Wf[c * NTGT + tg];
  atomicAdd(&out[(size_t)batch_seg[a] * NTGT + tg], acc);
}

// ================================================================
extern "C" void kernel_launch(void* const* d_in, const int* in_sizes, int n_in,
                              void* d_out, int out_size, void* d_ws, size_t ws_size,
                              hipStream_t stream) {
  // input order: setup_inputs() dict order, params expanded in insertion order
  const float* R            = (const float*)d_in[0];
  const float* p_freq       = (const float*)d_in[1];
  const float* p_emb        = (const float*)d_in[2];
  const float* p_emb_rbf_W  = (const float*)d_in[3];
  const float* p_emb_rbf_b  = (const float*)d_in[4];
  const float* p_emb_W      = (const float*)d_in[5];
  const float* p_emb_b      = (const float*)d_in[6];
  const float* p_out_rbf_W  = (const float*)d_in[7];
  const float* p_out_W      = (const float*)d_in[8];
  const float* p_out_b      = (const float*)d_in[9];
  const float* p_out_fin_W  = (const float*)d_in[10];
  const float* p_ji_W       = (const float*)d_in[11];
  const float* p_ji_b       = (const float*)d_in[12];
  const float* p_kj_W       = (const float*)d_in[13];
  const float* p_kj_b       = (const float*)d_in[14];
  const float* p_rbf_W      = (const float*)d_in[15];
  const float* p_sbf_W      = (const float*)d_in[16];
  const float* p_bilin      = (const float*)d_in[17];
  const float* p_bs_W       = (const float*)d_in[18];
  const float* p_bs_b       = (const float*)d_in[19];
  const float* p_fbs_W      = (const float*)d_in[20];
  const float* p_fbs_b      = (const float*)d_in[21];
  const float* p_as_W       = (const float*)d_in[22];
  const float* p_as_b       = (const float*)d_in[23];
  const int*   Z            = (const int*)d_in[24];
  const int*   idnb_i       = (const int*)d_in[25];
  const int*   idnb_j       = (const int*)d_in[26];
  const int*   id3_i        = (const int*)d_in[27];
  const int*   id3_j        = (const int*)d_in[28];
  const int*   id3_k        = (const int*)d_in[29];
  const int*   id_expand    = (const int*)d_in[30];
  const int*   id_reduce    = (const int*)d_in[31];
  const int*   batch_seg    = (const int*)d_in[32];

  const int E      = in_sizes[25];
  const int T      = in_sizes[27];
  const int natoms = in_sizes[24];
  float* outp = (float*)d_out;

  // --------- workspace bump allocator ---------
  char* base = (char*)d_ws;
  size_t off = 0;
  auto alloc = [&](size_t bytes) -> void* {
    void* p = base + off;
    off += (bytes + 255) & ~(size_t)255;
    return p;
  };
  float* normtab  = (float*)alloc(NSBF * sizeof(float));
  float* rbf      = (float*)alloc((size_t)E * NRAD * sizeof(float));
  float* rbf_env  = (float*)alloc((size_t)E * NSBF * sizeof(float));
  float* sbf      = (float*)alloc((size_t)T * NSBF * sizeof(float));
  float* sbf_p    = (float*)alloc((size_t)T * NBIL * sizeof(float));
  u16*   Aemb     = (u16*)alloc((size_t)E * 3 * EMB * sizeof(u16));
  float* x        = (float*)alloc((size_t)E * EMB * sizeof(float));
  u16*   xbf      = (u16*)alloc((size_t)E * EMB * sizeof(u16));
  float* x_ji     = (float*)alloc((size_t)E * EMB * sizeof(float));
  float* x_kj     = (float*)alloc((size_t)E * EMB * sizeof(float));
  u16*   xkj_bf   = (u16*)alloc((size_t)E * EMB * sizeof(u16));
  float* xkj_red  = (float*)alloc((size_t)E * EMB * sizeof(float));
  float* hbuf     = (float*)alloc((size_t)E * EMB * sizeof(float));
  u16*   hbf      = (u16*)alloc((size_t)E * EMB * sizeof(u16));
  float* x2       = (float*)alloc((size_t)E * EMB * sizeof(float));
  u16*   x2bf     = (u16*)alloc((size_t)E * EMB * sizeof(u16));
  float* t_atom   = (float*)alloc((size_t)natoms * EMB * sizeof(float));
  u16*   t_bf     = (u16*)alloc((size_t)natoms * EMB * sizeof(u16));
  float* t2       = (float*)alloc((size_t)natoms * EMB * sizeof(float));
  u16*   t2bf     = (u16*)alloc((size_t)natoms * EMB * sizeof(u16));
  u16*   wtbuf    = (u16*)alloc((size_t)3 * EMB * EMB * sizeof(u16)); // up to K=384
  u16*   bilinT   = (u16*)alloc((size_t)NBIL * EMB * EMB * sizeof(u16));
  (void)ws_size; (void)n_in;

  const int TPB = 256;
  auto blocks = [&](size_t n) { return (unsigned)((n + TPB - 1) / TPB); };
  const size_t Exf = (size_t)E * EMB;
  const size_t Axf = (size_t)natoms * EMB;

  auto zero = [&](float* p, size_t n) {
    k_zero_f32<<<blocks(n), TPB, 0, stream>>>(p, n);
  };
  auto tobf = [&](const float* a, u16* o, size_t n) {
    k_f32_to_bf16<<<blocks(n), TPB, 0, stream>>>(a, o, n);
  };
  auto pack = [&](const float* W, int K, int N) {
    k_pack_wt<<<blocks((size_t)K * N), TPB, 0, stream>>>(W, wtbuf, K, N);
  };
  auto gemm = [&](const u16* A, const float* bias, float* C, u16* Cbf,
                  int M, int K, int act, int acc) {
    unsigned mt = (unsigned)((M + 15) / 16);
    dim3 grid((mt + 7) / 8);
    k_wmma_gemm<<<grid, dim3(256), 0, stream>>>(A, wtbuf, bias, C, Cbf, M, K, act, acc);
  };

  // ---- output block (accumulates into d_out) ----
  auto out_block = [&](int bb) {
    zero(t_atom, Axf);
    k_out_gather<<<blocks(Exf), TPB, 0, stream>>>(
        rbf, p_out_rbf_W + (size_t)bb * NRAD * EMB, x, idnb_i, t_atom, E);
    tobf(t_atom, t_bf, Axf);
    float* cf = t_atom; u16* cb = t_bf; float* nf = t2; u16* nb = t2bf;
    for (int k = 0; k < 3; ++k) {
      pack(p_out_W + ((size_t)bb * 3 + k) * EMB * EMB, EMB, EMB);
      gemm(cb, p_out_b + ((size_t)bb * 3 + k) * EMB, nf, nb, natoms, EMB, 1, 0);
      float* tf = cf; cf = nf; nf = tf;
      u16* tb = cb; cb = nb; nb = tb;
    }
    k_final_out<<<blocks((size_t)natoms * NTGT), TPB, 0, stream>>>(
        cf, p_out_fin_W + (size_t)bb * EMB * NTGT, batch_seg, outp, natoms);
  };

  // ================= pipeline =================
  zero(outp, (size_t)out_size);
  k_init_norm<<<1, 64, 0, stream>>>(normtab);
  k_edge_basis<<<blocks(E), TPB, 0, stream>>>(R, idnb_i, idnb_j, p_freq, normtab,
                                              rbf, rbf_env, E);
  k_triplet_sbf<<<blocks(T), TPB, 0, stream>>>(R, id3_i, id3_j, id3_k, id_expand,
                                               rbf_env, sbf, T);
  // embedding
  k_embed_feat<<<blocks(Exf), TPB, 0, stream>>>(p_emb, Z, idnb_i, idnb_j, rbf,
                                                p_emb_rbf_W, p_emb_rbf_b, Aemb, E);
  pack(p_emb_W, 3 * EMB, EMB);
  gemm(Aemb, p_emb_b, x, xbf, E, 3 * EMB, 1, 0);

  out_block(0);

  for (int b = 0; b < NBLK; ++b) {
    // x_ji, x_kj
    pack(p_ji_W + (size_t)b * EMB * EMB, EMB, EMB);
    gemm(xbf, p_ji_b + (size_t)b * EMB, x_ji, nullptr, E, EMB, 1, 0);
    pack(p_kj_W + (size_t)b * EMB * EMB, EMB, EMB);
    gemm(xbf, p_kj_b + (size_t)b * EMB, x_kj, nullptr, E, EMB, 1, 0);
    k_kj_scale<<<blocks(Exf), TPB, 0, stream>>>(
        rbf, p_rbf_W + (size_t)b * NRAD * EMB, x_kj, xkj_bf, E);
    // triplet path
    k_sbf_project<<<blocks(T), TPB, 0, stream>>>(
        sbf, p_sbf_W + (size_t)b * NSBF * NBIL, sbf_p, T);
    k_pack_bilin<<<blocks((size_t)EMB * NBIL * EMB), TPB, 0, stream>>>(
        p_bilin + (size_t)b * EMB * NBIL * EMB, bilinT);
    zero(xkj_red, Exf);
    {
      unsigned wt = (unsigned)((T + 15) / 16);
      dim3 grid((wt + 7) / 8);
      k_wmma_bilinear<<<grid, dim3(256), 0, stream>>>(xkj_bf, sbf_p, bilinT,
                                                      id_expand, id_reduce,
                                                      xkj_red, T);
    }
    // x2 = x_ji + reduced
    k_add_bf<<<blocks(Exf), TPB, 0, stream>>>(x_ji, xkj_red, x2, x2bf, Exf);
    // before-skip residual (1 rep, 2 layers)
    pack(p_bs_W + ((size_t)b * 2 + 0) * EMB * EMB, EMB, EMB);
    gemm(x2bf, p_bs_b + ((size_t)b * 2 + 0) * EMB, hbuf, hbf, E, EMB, 1, 0);
    pack(p_bs_W + ((size_t)b * 2 + 1) * EMB * EMB, EMB, EMB);
    gemm(hbf, p_bs_b + ((size_t)b * 2 + 1) * EMB, x2, nullptr, E, EMB, 1, 1);
    tobf(x2, x2bf, Exf);
    // final-before-skip dense
    pack(p_fbs_W + (size_t)b * EMB * EMB, EMB, EMB);
    gemm(x2bf, p_fbs_b + (size_t)b * EMB, x2, nullptr, E, EMB, 1, 0);
    // x = x + x2
    k_accum_bf<<<blocks(Exf), TPB, 0, stream>>>(x, x2, xbf, Exf);
    // after-skip residuals (2 reps, 2 layers each)
    for (int r = 0; r < 2; ++r) {
      pack(p_as_W + (((size_t)b * 2 + r) * 2 + 0) * EMB * EMB, EMB, EMB);
      gemm(xbf, p_as_b + (((size_t)b * 2 + r) * 2 + 0) * EMB, hbuf, hbf, E, EMB, 1, 0);
      pack(p_as_W + (((size_t)b * 2 + r) * 2 + 1) * EMB * EMB, EMB, EMB);
      gemm(hbf, p_as_b + (((size_t)b * 2 + r) * 2 + 1) * EMB, x, nullptr, E, EMB, 1, 1);
      tobf(x, xbf, Exf);
    }
    out_block(b + 1);
  }
}
